// Flowtron_14267881357763
// MI455X (gfx1250) — matrix operand order repaired and checked by reading, not compiled, not run on hardware
//
#include <hip/hip_runtime.h>

// ============================================================================
// Flowtron on MI455X (gfx1250, wave32).
//
// Roofline: decoder = 2 flows x 512 sequential steps, ~27M MACs/step, every
// weight touched once per step -> weight-bandwidth bound. f32 weights/flow
// ~107MB (~= L2); bf16 conversion -> 53MB/flow, L2-resident at 192MB, halves
// traffic. Decoder matvecs: 64 WGs x 256 threads, 4 lanes/row, packed bf16x2
// loads, shfl_xor reduce, atomic grid barrier between dependent stages,
// GLOBAL_LOAD_ASYNC_TO_LDS_B128 for activation staging (ASYNCcnt-tracked).
// Encoder GEMMs (conv, Wih@X, K/V precompute) use v_wmma_f32_16x16x32_bf16
// with two independent accumulator chains to pipeline the XDL unit.
// ============================================================================

typedef __bf16 bh_t;
typedef __attribute__((ext_vector_type(16))) __bf16 v16bf;
typedef __attribute__((ext_vector_type(8)))  float  v8f;

#define TPB 256
#define NWG 64

#if defined(__AMDGCN__) && __has_builtin(__builtin_amdgcn_global_load_async_to_lds_b128)
#define USE_ASYNC_LDS 1
#else
#define USE_ASYNC_LDS 0
#endif

// builtin signature (from probe diagnostic): arg0 = int __vector_size__(16)
// in addrspace(1) (global), arg1 = LDS side, arg2/arg3 = imm offset / cpol.
typedef int v4i __attribute__((vector_size(16)));
typedef v4i __attribute__((address_space(1))) g_v4i;
typedef v4i __attribute__((address_space(3))) l_v4i;

__device__ __forceinline__ void wait_async0(){
#if defined(__AMDGCN__)
#if __has_builtin(__builtin_amdgcn_s_wait_asynccnt)
  __builtin_amdgcn_s_wait_asynccnt(0);
#else
  asm volatile("s_wait_asynccnt 0" ::: "memory");
#endif
#endif
}

__device__ __forceinline__ float sigf(float x){ return 1.0f/(1.0f+__expf(-x)); }

union FragAB { v16bf v; unsigned int u[8]; };
union FragC  { v8f  v; float f[8]; };

// ---------------- WMMA fragment loaders (contiguous K order, wave32) --------
// A 16x32 bf16: lane<16: M=lane, K=0..15; lane>=16: M=lane-16,
// K=16..31.  u[j] holds K = 2j + (lane>=16?16:0).
__device__ __forceinline__ v16bf load_a_frag(const bh_t* A, int lda, int mb,
                                             int kb, int lane){
  FragAB r;
  int m = mb + (lane & 15);
  int koff = (lane & 16) ? 16 : 0;
  const unsigned int* row = (const unsigned int*)(A + (size_t)m * lda);
#pragma unroll
  for (int j=0;j<8;++j){
    int k = kb + 2*j + koff;
    r.u[j] = row[k>>1];
  }
  return r.v;
}
// B 32x16 bf16: lane<16: N=lane, K=0..15; lane>=16: N=lane-16, K=16..31.
__device__ __forceinline__ v16bf load_b_frag(const unsigned short* B, int ldb,
                                             int nb, int kb, int lane){
  FragAB r;
  int n = nb + (lane & 15);
  int k0 = kb + ((lane & 16) ? 16 : 0);
#pragma unroll
  for (int j=0;j<8;++j){
    int k = k0 + 2*j;
    unsigned lo = B[(size_t)k*ldb + n];
    unsigned hi = B[(size_t)(k+1)*ldb + n];
    r.u[j] = lo | (hi<<16);
  }
  return r.v;
}

// ---------------- generic bf16 WMMA GEMM: D[M][N] = A[M][K]*B[K][N] (+bias) -
__global__ __launch_bounds__(TPB)
void gemm_bf16_wmma(const bh_t* __restrict__ A, const bh_t* __restrict__ B,
                    float* __restrict__ D, int M, int N, int K,
                    const float* __restrict__ bias){
  int wave = (blockIdx.x * TPB + threadIdx.x) >> 5;
  int lane = threadIdx.x & 31;
  int ntiles = N >> 4;
  if (wave >= (M >> 4) * ntiles) return;          // wave-uniform: EXEC all-1s
  int mb = (wave / ntiles) << 4;
  int nb = (wave % ntiles) << 4;
  v8f acc0 = {0.f,0.f,0.f,0.f,0.f,0.f,0.f,0.f};
  v8f acc1 = {0.f,0.f,0.f,0.f,0.f,0.f,0.f,0.f};
  const unsigned short* Bs = (const unsigned short*)B;
  int kb = 0;
  for (; kb + 64 <= K; kb += 64){                 // two independent chains
    v16bf a0 = load_a_frag(A, K, mb, kb,    lane);
    v16bf b0 = load_b_frag(Bs, N, nb, kb,   lane);
    v16bf a1 = load_a_frag(A, K, mb, kb+32, lane);
    v16bf b1 = load_b_frag(Bs, N, nb, kb+32,lane);
    acc0 = __builtin_amdgcn_wmma_f32_16x16x32_bf16(false, a0, false, b0,
                                                   (short)0, acc0, false, false);
    acc1 = __builtin_amdgcn_wmma_f32_16x16x32_bf16(false, a1, false, b1,
                                                   (short)0, acc1, false, false);
  }
  for (; kb < K; kb += 32){                       // K%64 tail (uniform)
    v16bf a0 = load_a_frag(A, K, mb, kb, lane);
    v16bf b0 = load_b_frag(Bs, N, nb, kb, lane);
    acc0 = __builtin_amdgcn_wmma_f32_16x16x32_bf16(false, a0, false, b0,
                                                   (short)0, acc0, false, false);
  }
  FragC c; c.v = acc0 + acc1;
  int n = nb + (lane & 15);
  int mo = (lane & 16) ? 8 : 0;
  if (bias != nullptr){                           // single uniform branch
#pragma unroll
    for (int r=0;r<8;++r){
      int m = mb + r + mo;
      D[(size_t)m*N + n] = c.f[r] + bias[m];
    }
  } else {
#pragma unroll
    for (int r=0;r<8;++r){
      int m = mb + r + mo;
      D[(size_t)m*N + n] = c.f[r];
    }
  }
}

// ---------------- conv1d(512->512,k=5,SAME) as 5 shifted WMMA GEMMs ---------
// Wp[k][o][i] bf16, xb padded [512][260] bf16 (p = t+2). y[o][t] f32 + bias.
__global__ __launch_bounds__(TPB)
void conv_wmma(const bh_t* __restrict__ Wp, const bh_t* __restrict__ xb,
               float* __restrict__ y, const float* __restrict__ bias){
  int wave = (blockIdx.x*TPB + threadIdx.x) >> 5;
  int lane = threadIdx.x & 31;
  if (wave >= 512) return;
  int mb = (wave >> 4) << 4;
  int nb = (wave & 15) << 4;
  v8f acc0 = {0.f,0.f,0.f,0.f,0.f,0.f,0.f,0.f};
  v8f acc1 = {0.f,0.f,0.f,0.f,0.f,0.f,0.f,0.f};
  const unsigned short* xs = (const unsigned short*)xb;
  for (int k0=0;k0<5;++k0){
    const bh_t* A = Wp + (size_t)k0*512*512;
    int n = nb + (lane&15) + k0;                  // shifted padded-time column
    for (int kb=0; kb<512; kb+=64){
      v16bf a0 = load_a_frag(A, 512, mb, kb,    lane);
      v16bf a1 = load_a_frag(A, 512, mb, kb+32, lane);
      FragAB rb0, rb1;
      int kk0 = kb + ((lane&16)?16:0);
#pragma unroll
      for (int j=0;j<8;++j){
        int k = kk0 + 2*j;
        rb0.u[j] = (unsigned)xs[(size_t)k*260 + n]
                 | ((unsigned)xs[(size_t)(k+1)*260 + n] << 16);
        rb1.u[j] = (unsigned)xs[(size_t)(k+32)*260 + n]
                 | ((unsigned)xs[(size_t)(k+33)*260 + n] << 16);
      }
      acc0 = __builtin_amdgcn_wmma_f32_16x16x32_bf16(false, a0, false, rb0.v,
                                                     (short)0, acc0, false, false);
      acc1 = __builtin_amdgcn_wmma_f32_16x16x32_bf16(false, a1, false, rb1.v,
                                                     (short)0, acc1, false, false);
    }
  }
  FragC c; c.v = acc0 + acc1;
  int n = nb + (lane & 15);
  int mo = (lane & 16) ? 8 : 0;
#pragma unroll
  for (int r=0;r<8;++r){
    int m = mb + r + mo;
    y[(size_t)m*256 + n] = c.f[r] + bias[m];
  }
}

// ---------------- small elementwise / packing kernels -----------------------
__global__ void embed_kernel(const int* __restrict__ text,
                             const float* __restrict__ emb,
                             float* __restrict__ x){
  int i = blockIdx.x*blockDim.x + threadIdx.x;
  if (i < 512*256){ int c=i>>8, t=i&255; x[i] = emb[(size_t)text[t]*512 + c]; }
}

__global__ void pack_convw(const float* __restrict__ w, bh_t* __restrict__ wp){
  int i = blockIdx.x*blockDim.x + threadIdx.x;           // wp[k][o][i]
  if (i < 5*512*512){
    int k = i/(512*512); int r = i - k*512*512; int o = r>>9; int ii = r&511;
    wp[i] = (bh_t)w[((size_t)o*512 + ii)*5 + k];
  }
}

__global__ void build_xb(const float* __restrict__ x, bh_t* __restrict__ xb){
  int i = blockIdx.x*blockDim.x + threadIdx.x;           // [512][260] padded
  if (i < 512*260){
    int c = i/260, p = i - c*260;
    xb[i] = (p>=2 && p<258) ? (bh_t)x[c*256 + (p-2)] : (bh_t)0.0f;
  }
}

__global__ void f2b(const float* __restrict__ s, bh_t* __restrict__ d, int n){
  int i = blockIdx.x*blockDim.x + threadIdx.x;
  if (i < n) d[i] = (bh_t)s[i];
}

__global__ __launch_bounds__(256)
void inorm_relu(float* __restrict__ x, const float* __restrict__ w,
                const float* __restrict__ b){
  int c = blockIdx.x, tid = threadIdx.x;
  __shared__ float red[256];
  float v = x[(size_t)c*256 + tid];
  red[tid] = v; __syncthreads();
  for (int o=128;o;o>>=1){ if (tid<o) red[tid]+=red[tid+o]; __syncthreads(); }
  float mean = red[0]/256.f; __syncthreads();
  float d = v - mean;
  red[tid] = d*d; __syncthreads();
  for (int o=128;o;o>>=1){ if (tid<o) red[tid]+=red[tid+o]; __syncthreads(); }
  float sd = sqrtf(red[0]/255.f);                        // ddof=1
  float xn = d/(sd+1e-8f)*w[c] + b[c];
  x[(size_t)c*256 + tid] = fmaxf(xn, 0.f);
}

// encoder biLSTM (hid=256): gates_in precomputed by WMMA GEMM; sequential part
// keeps h,c in LDS; block 0 = fwd, block 1 = bwd. Writes encTf[j][t] (f32).
__global__ __launch_bounds__(256)
void enc_lstm_seq(const float* __restrict__ gf, const float* __restrict__ gb,
                  const float* __restrict__ WhhF, const float* __restrict__ WhhB,
                  const float* __restrict__ bihF, const float* __restrict__ bhhF,
                  const float* __restrict__ bihB, const float* __restrict__ bhhB,
                  float* __restrict__ encTf){
  int tid = threadIdx.x; bool bwd = (blockIdx.x != 0);
  const float* gin = bwd ? gb   : gf;
  const float* Whh = bwd ? WhhB : WhhF;
  const float* b0  = bwd ? bihB : bihF;
  const float* b1  = bwd ? bhhB : bhhF;
  __shared__ float h[256], c[256];
  h[tid]=0.f; c[tid]=0.f; __syncthreads();
  for (int s=0;s<256;++s){
    int t = bwd ? 255-s : s;
    float g4[4];
#pragma unroll
    for (int j=0;j<4;++j){
      int r = tid + j*256;
      float acc = gin[(size_t)r*256 + t] + b0[r] + b1[r];
      const float* Wr = Whh + (size_t)r*256;
      for (int k=0;k<256;++k) acc = fmaf(Wr[k], h[k], acc);
      g4[j] = acc;
    }
    __syncthreads();                                     // gate reads of h done
    float ii=sigf(g4[0]), ff=sigf(g4[1]), gg=tanhf(g4[2]), oo=sigf(g4[3]);
    float cc = ff*c[tid] + ii*gg; c[tid]=cc;
    float hh = oo*tanhf(cc);
    h[tid]=hh;
    encTf[(size_t)((bwd?256:0)+tid)*256 + t] = hh;
    __syncthreads();
  }
}

__global__ void spk_fill(const float* __restrict__ spk_emb,
                         const int* __restrict__ spkv,
                         float* __restrict__ encTf){
  int i = blockIdx.x*blockDim.x + threadIdx.x;
  if (i < 128*256){
    int s = i>>8, t = i&255;
    encTf[(size_t)(512+s)*256 + t] = spk_emb[(size_t)spkv[0]*128 + s];
  }
}

__global__ void tr_res(const float* __restrict__ r, float* __restrict__ o){
  int i = blockIdx.x*blockDim.x + threadIdx.x;           // [80][512]->[512][80]
  if (i < 80*512){ int m=i/512, t=i-m*512; o[(size_t)t*80+m] = r[i]; }
}

__global__ void final_out(const float* __restrict__ r, float* __restrict__ o){
  int i = blockIdx.x*blockDim.x + threadIdx.x;           // [512][80]->[80][512]
  if (i < 80*512){ int m=i/512, t=i-m*512; o[i] = r[(size_t)t*80+m]; }
}

__global__ void init_bar(unsigned int* bar){ bar[0]=0u; bar[1]=0u; }

// ---------------- AR decoder: persistent grid, atomic sense barrier ---------
struct DecW {
  const unsigned int *attWih, *attWhh, *Wq, *l0Wih, *l0Whh, *l1Wih, *l1Whh,
                     *dw1, *dw2, *cw;                    // packed bf16x2
  const float *att_bih, *att_bhh, *l0_bih, *l0_bhh, *l1_bih, *l1_bhh;
  const float *db1, *db2, *cb, *vatt;
  const float *k_at, *vv_at;                             // [640][256] f32
};
struct DecS {
  float *h_att, *c_att, *h10, *c10, *h11, *c11;
  float *prev, *gates, *ctx, *d1, *d2;
  unsigned int* bar;
};

// global f32 -> LDS staging. n must be a multiple of 4 (all call sites are).
// CDNA5 path: GLOBAL_LOAD_ASYNC_TO_LDS_B128, tracked by ASYNCcnt; fallback is
// a plain VGPR round-trip copy. Caller must __syncthreads() after.
__device__ __forceinline__ void stage_to_lds(float* __restrict__ dst,
                                             const float* __restrict__ src,
                                             int n, int tid){
#if USE_ASYNC_LDS
  for (int i = tid*4; i < n; i += TPB*4){
    __builtin_amdgcn_global_load_async_to_lds_b128(
        (g_v4i*)(src + i), (l_v4i*)(dst + i), 0, 0);
  }
  wait_async0();
#else
  for (int i = tid; i < n; i += TPB) dst[i] = src[i];
#endif
}

__device__ __forceinline__ float rowdot4(const unsigned int* Wrow, int K,
                                         const float* x, int sub){
  float acc = 0.f;
  for (int k = sub*2; k < K; k += 8){
    unsigned int p = Wrow[k>>1];
    __builtin_prefetch(&Wrow[(k>>1) + 64], 0, 1);        // global_prefetch_b8
    acc = fmaf(__uint_as_float(p<<16),          x[k],   acc);
    acc = fmaf(__uint_as_float(p&0xffff0000u),  x[k+1], acc);
  }
  return acc;
}
__device__ __forceinline__ float rowdot1(const unsigned int* Wrow, int K,
                                         const float* x){
  float acc = 0.f;
  for (int k = 0; k < K; k += 2){
    unsigned int p = Wrow[k>>1];
    acc = fmaf(__uint_as_float(p<<16),         x[k],   acc);
    acc = fmaf(__uint_as_float(p&0xffff0000u), x[k+1], acc);
  }
  return acc;
}

__device__ void gbar(unsigned int* bar){
  __threadfence();
  __syncthreads();
  if (threadIdx.x == 0){
    unsigned g = __hip_atomic_load(&bar[1], __ATOMIC_RELAXED, __HIP_MEMORY_SCOPE_AGENT);
    unsigned a = __hip_atomic_fetch_add(&bar[0], 1u, __ATOMIC_ACQ_REL, __HIP_MEMORY_SCOPE_AGENT);
    if (a == NWG-1){
      __hip_atomic_store(&bar[0], 0u, __ATOMIC_RELAXED, __HIP_MEMORY_SCOPE_AGENT);
      __hip_atomic_fetch_add(&bar[1], 1u, __ATOMIC_RELEASE, __HIP_MEMORY_SCOPE_AGENT);
    } else {
      while (__hip_atomic_load(&bar[1], __ATOMIC_ACQUIRE, __HIP_MEMORY_SCOPE_AGENT) == g)
        __builtin_amdgcn_s_sleep(2);
    }
  }
  __syncthreads();
}

// 4096 gate rows, 64 rows/WG, 4 lanes/row, two K-segments + biases.
__device__ void big_gates(const unsigned int* Wa, int Ka, const float* xa,
                          const unsigned int* Wb, int Kb, const float* xb_,
                          const float* b0, const float* b1, float* out,
                          int wg, int tid){
  int row = wg*64 + (tid>>2);
  int sub = tid & 3;
  float acc = rowdot4(Wa + (size_t)row*(Ka>>1), Ka, xa,  sub)
            + rowdot4(Wb + (size_t)row*(Kb>>1), Kb, xb_, sub);
  acc += __shfl_xor(acc, 1);
  acc += __shfl_xor(acc, 2);
  if (sub == 0) out[row] = acc + b0[row] + b1[row];
}

__device__ void cell_update(const float* g, float* h, float* c, int gtid){
  if (gtid < 1024){
    float ii = sigf(g[gtid]);
    float ff = sigf(g[gtid+1024]);
    float gg = tanhf(g[gtid+2048]);
    float oo = sigf(g[gtid+3072]);
    float cc = ff*c[gtid] + ii*gg;
    c[gtid] = cc;
    h[gtid] = oo*tanhf(cc);
  }
}

__global__ __launch_bounds__(TPB)
void decoder_kernel(DecW w, DecS s, const float* __restrict__ res_in,
                    float* __restrict__ res_out, int reverse){
  __shared__ float lds[2816];
  int tid = threadIdx.x, wg = blockIdx.x;
  int gtid = wg*TPB + tid;
  if (gtid < 1024){
    s.h_att[gtid]=0.f; s.c_att[gtid]=0.f; s.h10[gtid]=0.f;
    s.c10[gtid]=0.f;   s.h11[gtid]=0.f;   s.c11[gtid]=0.f;
  }
  if (gtid < 80) s.prev[gtid]=0.f;
  gbar(s.bar);

  for (int t=0; t<512; ++t){
    // ---- attention LSTM gates: K = 80 (prev out) + 1024 (h_att)
    stage_to_lds(lds,      s.h_att, 1024, tid);
    stage_to_lds(lds+1024, s.prev,  80,   tid);
    __syncthreads();
    big_gates(w.attWih, 80, lds+1024, w.attWhh, 1024, lds,
              w.att_bih, w.att_bhh, s.gates, wg, tid);
    gbar(s.bar);
    cell_update(s.gates, s.h_att, s.c_att, gtid);
    gbar(s.bar);

    // ---- WG0: q = Wq@h, scores/softmax/ctx (small: 1.2M MAC total)
    if (wg == 0){
      stage_to_lds(lds, s.h_att, 1024, tid);
      __syncthreads();
      for (int r=tid;r<640;r+=TPB)                       // q -> lds[1024+r]
        lds[1024+r] = rowdot1(w.Wq + (size_t)r*512, 1024, lds);
      __syncthreads();
      float sc = 0.f;
      for (int a=0;a<640;++a)
        sc += tanhf(lds[1024+a] + w.k_at[(size_t)a*256 + tid]) * w.vatt[a];
      lds[1664+tid] = sc; __syncthreads();
      for (int o=128;o;o>>=1){ if(tid<o) lds[1664+tid]=fmaxf(lds[1664+tid],lds[1664+tid+o]); __syncthreads(); }
      float mx = lds[1664]; __syncthreads();
      float e = __expf(sc - mx);
      lds[1664+tid] = e; __syncthreads();
      for (int o=128;o;o>>=1){ if(tid<o) lds[1664+tid]+=lds[1664+tid+o]; __syncthreads(); }
      float inv = 1.f/lds[1664]; __syncthreads();
      lds[1664+tid] = e*inv;                             // attn weights
      __syncthreads();
      for (int a=tid;a<640;a+=TPB){
        const float* vva = w.vv_at + (size_t)a*256;
        float acc = 0.f;
        for (int ss=0;ss<256;++ss) acc = fmaf(lds[1664+ss], vva[ss], acc);
        s.ctx[a] = acc;
      }
    }
    gbar(s.bar);

    // ---- LSTM0 gates: K = 1664 (dec_in = h_att|ctx) + 1024 (h10)
    stage_to_lds(lds,      s.h_att, 1024, tid);
    stage_to_lds(lds+1024, s.ctx,   640,  tid);
    stage_to_lds(lds+1664, s.h10,   1024, tid);
    __syncthreads();
    big_gates(w.l0Wih, 1664, lds, w.l0Whh, 1024, lds+1664,
              w.l0_bih, w.l0_bhh, s.gates, wg, tid);
    gbar(s.bar);
    cell_update(s.gates, s.h10, s.c10, gtid);
    gbar(s.bar);

    // ---- LSTM1 gates: K = 1024 (h10) + 1024 (h11)
    stage_to_lds(lds,      s.h10, 1024, tid);
    stage_to_lds(lds+1024, s.h11, 1024, tid);
    __syncthreads();
    big_gates(w.l1Wih, 1024, lds, w.l1Whh, 1024, lds+1024,
              w.l1_bih, w.l1_bhh, s.gates, wg, tid);
    gbar(s.bar);
    cell_update(s.gates, s.h11, s.c11, gtid);
    gbar(s.bar);

    // ---- dense1 (tanh), rows 1024, WGs 0..15
    {
      stage_to_lds(lds, s.h11, 1024, tid);
      __syncthreads();
      int row = gtid>>2, sub = gtid&3;
      if (row < 1024){
        float acc = rowdot4(w.dw1 + (size_t)row*512, 1024, lds, sub);
        acc += __shfl_xor(acc,1); acc += __shfl_xor(acc,2);
        if (sub==0) s.d1[row] = tanhf(acc + w.db1[row]);
      }
    }
    gbar(s.bar);
    // ---- dense2 (tanh)
    {
      stage_to_lds(lds, s.d1, 1024, tid);
      __syncthreads();
      int row = gtid>>2, sub = gtid&3;
      if (row < 1024){
        float acc = rowdot4(w.dw2 + (size_t)row*512, 1024, lds, sub);
        acc += __shfl_xor(acc,1); acc += __shfl_xor(acc,2);
        if (sub==0) s.d2[row] = tanhf(acc + w.db2[row]);
      }
    }
    gbar(s.bar);
    // ---- conv 160x1024 + affine-flow output, WG0
    if (wg == 0){
      stage_to_lds(lds, s.d2, 1024, tid);
      __syncthreads();
      if (tid < 160)
        lds[1024+tid] = rowdot1(w.cw + (size_t)tid*512, 1024, lds) + w.cb[tid];
      __syncthreads();
      if (tid < 80){
        int tt = reverse ? (511-t) : t;
        float r  = res_in[(size_t)tt*80 + tid];
        float ls = lds[1024+tid];
        float bb = lds[1104+tid];
        float o  = (r - bb) * __expf(-ls);
        s.prev[tid] = o;
        res_out[(size_t)tt*80 + tid] = o;
      }
    }
    gbar(s.bar);
  }
}

// ============================================================================
// Host
// ============================================================================
extern "C" void kernel_launch(void* const* d_in, const int* in_sizes, int n_in,
                              void* d_out, int out_size, void* d_ws, size_t ws_size,
                              hipStream_t stream){
  (void)in_sizes; (void)n_in; (void)out_size; (void)ws_size;
  auto F = [&](int i){ return (const float*)d_in[i]; };

  const float* residual = (const float*)d_in[0];   // [80][512]
  const int*   spkv     = (const int*)  d_in[1];
  const int*   text     = (const int*)  d_in[2];
  const float* spk_emb  = F(3);
  const float* txt_emb  = F(4);
  // enc convs: 5..16, enc lstm fwd 17..20, bwd 21..24, flows start 25 (22 each)

  char* ws = (char*)d_ws; size_t off = 0;
  auto alloc = [&](size_t b)->void*{
    void* p = ws + off; off += (b + 255) & ~(size_t)255; return p;
  };
  unsigned int* bar = (unsigned int*)alloc(256);
  float* x0    = (float*)alloc(512*256*4);
  float* x1    = (float*)alloc(512*256*4);
  bh_t*  xb    = (bh_t*) alloc(512*260*2);
  bh_t*  wp    = (bh_t*) alloc((size_t)5*512*512*2);
  bh_t*  xc    = (bh_t*) alloc(512*256*2);
  bh_t*  wihF  = (bh_t*) alloc((size_t)1024*512*2);
  bh_t*  wihB  = (bh_t*) alloc((size_t)1024*512*2);
  float* gf    = (float*)alloc((size_t)1024*256*4);
  float* gb    = (float*)alloc((size_t)1024*256*4);
  float* encTf = (float*)alloc((size_t)640*256*4);
  bh_t*  encTb = (bh_t*) alloc((size_t)640*256*2);
  bh_t*  wkB   = (bh_t*) alloc((size_t)640*640*2);
  bh_t*  wvB   = (bh_t*) alloc((size_t)640*640*2);
  float* k_at  = (float*)alloc((size_t)640*256*4);
  float* vv_at = (float*)alloc((size_t)640*256*4);
  // decoder bf16 weights (reused by both flows; flows run sequentially)
  bh_t* attWih = (bh_t*)alloc((size_t)4096*80*2);
  bh_t* attWhh = (bh_t*)alloc((size_t)4096*1024*2);
  bh_t* wq     = (bh_t*)alloc((size_t)640*1024*2);
  bh_t* l0Wih  = (bh_t*)alloc((size_t)4096*1664*2);
  bh_t* l0Whh  = (bh_t*)alloc((size_t)4096*1024*2);
  bh_t* l1Wih  = (bh_t*)alloc((size_t)4096*1024*2);
  bh_t* l1Whh  = (bh_t*)alloc((size_t)4096*1024*2);
  bh_t* dw1    = (bh_t*)alloc((size_t)1024*1024*2);
  bh_t* dw2    = (bh_t*)alloc((size_t)1024*1024*2);
  bh_t* cw     = (bh_t*)alloc((size_t)160*1024*2);
  // decoder state + residual chain
  float* h_att=(float*)alloc(1024*4); float* c_att=(float*)alloc(1024*4);
  float* h10  =(float*)alloc(1024*4); float* c10  =(float*)alloc(1024*4);
  float* h11  =(float*)alloc(1024*4); float* c11  =(float*)alloc(1024*4);
  float* prev =(float*)alloc(128*4);  float* gates=(float*)alloc(4096*4);
  float* ctx  =(float*)alloc(640*4);
  float* dd1  =(float*)alloc(1024*4); float* dd2  =(float*)alloc(1024*4);
  float* res0 =(float*)alloc((size_t)512*80*4);
  float* res1 =(float*)alloc((size_t)512*80*4);
  float* res2 =(float*)alloc((size_t)512*80*4);

  auto blocks = [](long n){ return (int)((n + TPB - 1) / TPB); };
  auto cvt = [&](const float* s, bh_t* d, long n){
    f2b<<<blocks(n), TPB, 0, stream>>>(s, d, (int)n);
  };
  auto gemm = [&](const bh_t* A, const bh_t* B, float* D, int M, int N, int K,
                  const float* bias){
    int waves = (M/16)*(N/16);
    gemm_bf16_wmma<<<blocks((long)waves*32), TPB, 0, stream>>>(A,B,D,M,N,K,bias);
  };

  // ---- text encoder ----
  embed_kernel<<<blocks(512*256), TPB, 0, stream>>>(text, txt_emb, x0);
  float* cur = x0; float* oth = x1;
  for (int l=0;l<3;++l){
    const float* conv_w = F(5 + 4*l);
    const float* conv_b = F(6 + 4*l);
    const float* in_w   = F(7 + 4*l);
    const float* in_b   = F(8 + 4*l);
    pack_convw<<<blocks((long)5*512*512), TPB, 0, stream>>>(conv_w, wp);
    build_xb  <<<blocks(512*260), TPB, 0, stream>>>(cur, xb);
    conv_wmma <<<64, TPB, 0, stream>>>(wp, xb, oth, conv_b);
    inorm_relu<<<512, 256, 0, stream>>>(oth, in_w, in_b);
    float* t2 = cur; cur = oth; oth = t2;
  }
  cvt(cur, (bh_t*)xc, 512*256);
  cvt(F(17), wihF, (long)1024*512);
  cvt(F(21), wihB, (long)1024*512);
  gemm(wihF, xc, gf, 1024, 256, 512, nullptr);           // Wih @ X (fwd)
  gemm(wihB, xc, gb, 1024, 256, 512, nullptr);           // Wih @ X (bwd)
  enc_lstm_seq<<<2, 256, 0, stream>>>(gf, gb, F(18), F(22),
                                      F(19), F(20), F(23), F(24), encTf);
  spk_fill<<<blocks(128*256), TPB, 0, stream>>>(spk_emb, spkv, encTf);
  cvt(encTf, encTb, (long)640*256);

  // ---- residual transpose ----
  tr_res<<<blocks(80*512), TPB, 0, stream>>>(residual, res0);

  // ---- flows: i = 1 (reversed), then i = 0 ----
  for (int fi = 1; fi >= 0; --fi){
    int fb = 25 + 22*fi;
    cvt(F(fb+0),  attWih, (long)4096*80);
    cvt(F(fb+1),  attWhh, (long)4096*1024);
    cvt(F(fb+4),  wq,     (long)640*1024);
    cvt(F(fb+8),  l0Wih,  (long)4096*1664);
    cvt(F(fb+9),  l0Whh,  (long)4096*1024);
    cvt(F(fb+12), l1Wih,  (long)4096*1024);
    cvt(F(fb+13), l1Whh,  (long)4096*1024);
    cvt(F(fb+16), dw1,    (long)1024*1024);
    cvt(F(fb+18), dw2,    (long)1024*1024);
    cvt(F(fb+20), cw,     (long)160*1024);
    cvt(F(fb+5),  wkB,    (long)640*640);
    cvt(F(fb+6),  wvB,    (long)640*640);
    gemm(wkB, encTb, k_at,  640, 256, 640, nullptr);     // K  precompute [a][t]
    gemm(wvB, encTb, vv_at, 640, 256, 640, nullptr);     // V  precompute [a][t]

    DecW w;
    w.attWih=(const unsigned int*)attWih; w.attWhh=(const unsigned int*)attWhh;
    w.Wq=(const unsigned int*)wq;
    w.l0Wih=(const unsigned int*)l0Wih;   w.l0Whh=(const unsigned int*)l0Whh;
    w.l1Wih=(const unsigned int*)l1Wih;   w.l1Whh=(const unsigned int*)l1Whh;
    w.dw1=(const unsigned int*)dw1; w.dw2=(const unsigned int*)dw2;
    w.cw=(const unsigned int*)cw;
    w.att_bih=F(fb+2);  w.att_bhh=F(fb+3);
    w.l0_bih=F(fb+10);  w.l0_bhh=F(fb+11);
    w.l1_bih=F(fb+14);  w.l1_bhh=F(fb+15);
    w.db1=F(fb+17); w.db2=F(fb+19); w.cb=F(fb+21); w.vatt=F(fb+7);
    w.k_at=k_at; w.vv_at=vv_at;

    DecS s;
    s.h_att=h_att; s.c_att=c_att; s.h10=h10; s.c10=c10; s.h11=h11; s.c11=c11;
    s.prev=prev; s.gates=gates; s.ctx=ctx; s.d1=dd1; s.d2=dd2; s.bar=bar;

    const float* rin  = (fi==1) ? res0 : res1;
    float*       rout = (fi==1) ? res1 : res2;
    init_bar<<<1, 1, 0, stream>>>(bar);
    decoder_kernel<<<NWG, TPB, 0, stream>>>(w, s, rin, rout, fi==1 ? 1 : 0);
  }

  final_out<<<blocks(80*512), TPB, 0, stream>>>(res2, (float*)d_out);
}
